// LSTMDecoder_9663676416058
// MI455X (gfx1250) — matrix operand order repaired and checked by reading, not compiled, not run on hardware
//
#include <hip/hip_runtime.h>
#include <hip/hip_bf16.h>

// Problem dims
#define Bn   64
#define Nn   196
#define ENCn 2048
#define Dn   512
#define An   512
#define En   512
#define Vn   32000
#define Tn   20
#define Sn   19          // scan steps = T-1
#define KG   3072        // gates K = E + ENC + D
#define KO   2560        // out-head K = D + ENC

typedef __bf16 bf16;
typedef bf16  bf16x8  __attribute__((ext_vector_type(8)));
typedef bf16  bf16x16 __attribute__((ext_vector_type(16)));
typedef float f32x8   __attribute__((ext_vector_type(8)));

__device__ __forceinline__ float sigm(float x) { return 1.0f / (1.0f + __expf(-x)); }

// ---- fragment loaders -----------------------------------------------------
// ISA 16-bit A layout: lanes 0-15 hold K {0..7,16..23}, lanes 16-31 {8..15,24..31}
template <typename AT>
__device__ __forceinline__ bf16x16 load_a_frag(const AT* __restrict__ arow, int kb, int half) {
    if constexpr (sizeof(AT) == 4) {
        f32x8 a0 = *(const f32x8*)(arow + kb + half * 8);
        f32x8 a1 = *(const f32x8*)(arow + kb + 16 + half * 8);
        bf16x16 af;
        #pragma unroll
        for (int i = 0; i < 8; ++i) { af[i] = (bf16)a0[i]; af[i + 8] = (bf16)a1[i]; }
        return af;
    } else {
        bf16x8 a0 = *(const bf16x8*)(arow + kb + half * 8);
        bf16x8 a1 = *(const bf16x8*)(arow + kb + 16 + half * 8);
        return __builtin_shufflevector(a0, a1, 0,1,2,3,4,5,6,7,8,9,10,11,12,13,14,15);
    }
}
// B layout matches A: lanes 0-15 hold K {0..7,16..23}, lanes 16-31 {8..15,24..31}
__device__ __forceinline__ bf16x16 load_b_frag(const bf16* __restrict__ brow, int kb, int half) {
    bf16x8 b0 = *(const bf16x8*)(brow + kb + half * 8);
    bf16x8 b1 = *(const bf16x8*)(brow + kb + 16 + half * 8);
    return __builtin_shufflevector(b0, b1, 0,1,2,3,4,5,6,7,8,9,10,11,12,13,14,15);
}

// ---------------------------------------------------------------------------
// WMMA GEMM:  C[M,N] = A[M,K] * W[N,K]^T (+bias[n]) (+add[m,n])
// Each wave computes a 16 x (16*NT) strip of C. The A fragment is loaded once
// per 32-wide K chunk and reused across NT back-to-back v_wmma issues.
// The K loop is software-pipelined over chunk PAIRS with two alternating
// register buffer sets, so load latency overlaps WMMA work in-wave and no
// rotating-buffer register copies are needed. Requires K % 64 == 0.
// ---------------------------------------------------------------------------
template <typename AT, int NT>
__global__ __launch_bounds__(128) void wmma_gemm(
    const AT* __restrict__ A, int lda,
    const bf16* __restrict__ Bw, int ldb,
    const float* __restrict__ bias,
    const float* __restrict__ add, int ldadd,
    float* __restrict__ C, int ldc,
    bf16* __restrict__ Cb, int ldcb,
    int M, int N, int K)
{
    const int lane   = threadIdx.x & 31;
    const int wid    = blockIdx.x * (blockDim.x >> 5) + (threadIdx.x >> 5);
    const int groupsN = N / (16 * NT);
    if (wid >= (M >> 4) * groupsN) return;
    const int tm   = wid / groupsN;
    const int tng  = wid % groupsN;
    const int half = lane >> 4;
    const int l15  = lane & 15;

    const AT* arow = A + (size_t)(tm * 16 + l15) * lda;
    const bf16* brow[NT];
    #pragma unroll
    for (int t = 0; t < NT; ++t)
        brow[t] = Bw + (size_t)((tng * NT + t) * 16 + l15) * ldb;

    f32x8 acc[NT];
    #pragma unroll
    for (int t = 0; t < NT; ++t) acc[t] = (f32x8){0.f,0.f,0.f,0.f,0.f,0.f,0.f,0.f};

    // prologue: fetch chunk 0 into buffer set 0
    bf16x16 a0 = load_a_frag<AT>(arow, 0, half);
    bf16x16 b0[NT];
    #pragma unroll
    for (int t = 0; t < NT; ++t) b0[t] = load_b_frag(brow[t], 0, half);

    bf16x16 a1;
    bf16x16 b1[NT];
    int kb = 0;
    for (; kb + 64 < K; kb += 64) {
        // fetch chunk kb+32 into set 1, consume set 0
        a1 = load_a_frag<AT>(arow, kb + 32, half);
        #pragma unroll
        for (int t = 0; t < NT; ++t) b1[t] = load_b_frag(brow[t], kb + 32, half);
        #pragma unroll
        for (int t = 0; t < NT; ++t)
            acc[t] = __builtin_amdgcn_wmma_f32_16x16x32_bf16(false, a0, false, b0[t],
                                                             (short)0, acc[t], false, false);
        // fetch chunk kb+64 into set 0, consume set 1
        a0 = load_a_frag<AT>(arow, kb + 64, half);
        #pragma unroll
        for (int t = 0; t < NT; ++t) b0[t] = load_b_frag(brow[t], kb + 64, half);
        #pragma unroll
        for (int t = 0; t < NT; ++t)
            acc[t] = __builtin_amdgcn_wmma_f32_16x16x32_bf16(false, a1, false, b1[t],
                                                             (short)0, acc[t], false, false);
    }
    // final pair: chunk kb already in set 0, fetch kb+32 then drain both
    a1 = load_a_frag<AT>(arow, kb + 32, half);
    #pragma unroll
    for (int t = 0; t < NT; ++t) b1[t] = load_b_frag(brow[t], kb + 32, half);
    #pragma unroll
    for (int t = 0; t < NT; ++t)
        acc[t] = __builtin_amdgcn_wmma_f32_16x16x32_bf16(false, a0, false, b0[t],
                                                         (short)0, acc[t], false, false);
    #pragma unroll
    for (int t = 0; t < NT; ++t)
        acc[t] = __builtin_amdgcn_wmma_f32_16x16x32_bf16(false, a1, false, b1[t],
                                                         (short)0, acc[t], false, false);

    #pragma unroll
    for (int t = 0; t < NT; ++t) {
        const int   n  = (tng * NT + t) * 16 + l15;
        const float bv = bias ? bias[n] : 0.0f;
        #pragma unroll
        for (int r = 0; r < 8; ++r) {
            const int m = tm * 16 + half * 8 + r;   // ISA C/D layout
            float v = acc[t][r] + bv;
            if (add) v += add[(size_t)m * ldadd + n];
            if (C)   C[(size_t)m * ldc + n] = v;
            if (Cb)  Cb[(size_t)m * ldcb + n] = (bf16)v;
        }
    }
}

// ---------------------------------------------------------------------------
// Setup kernels
// ---------------------------------------------------------------------------
__global__ void cvt_f32_bf16(const float* __restrict__ s, bf16* __restrict__ d, int n) {
    int i = blockIdx.x * blockDim.x + threadIdx.x;
    if (i < n) d[i] = (bf16)s[i];
}

__global__ void concat_gates_w(const float* __restrict__ Wih, const float* __restrict__ Whh,
                               bf16* __restrict__ d) {
    int i = blockIdx.x * blockDim.x + threadIdx.x;     // over 2048*3072
    if (i >= 2048 * KG) return;
    int r = i / KG, k = i % KG;
    float v = (k < 2560) ? Wih[r * 2560 + k] : Whh[r * Dn + (k - 2560)];
    d[i] = (bf16)v;
}

__global__ void concat_out_w(const float* __restrict__ hidW, const float* __restrict__ ctxW,
                             bf16* __restrict__ d) {
    int i = blockIdx.x * blockDim.x + threadIdx.x;     // over 512*2560
    if (i >= En * KO) return;
    int r = i / KO, k = i % KO;
    float v = (k < Dn) ? hidW[r * Dn + k] : ctxW[r * ENCn + (k - Dn)];
    d[i] = (bf16)v;
}

__global__ void add_bias(const float* __restrict__ a, const float* __restrict__ b,
                         float* __restrict__ d, int n) {
    int i = blockIdx.x * blockDim.x + threadIdx.x;
    if (i < n) d[i] = a[i] + b[i];
}

// gather e_t = emb[cap[b, s]]  ->  fp32 residual buffer + bf16 into xh[:, 0:512]
__global__ void gather_emb(const int* __restrict__ cap, const float* __restrict__ emb,
                           int s, float* __restrict__ et, bf16* __restrict__ xh) {
    int i = blockIdx.x * blockDim.x + threadIdx.x;     // Bn*En
    if (i >= Bn * En) return;
    int b = i >> 9, j = i & 511;
    float v = emb[(size_t)cap[b * Tn + s] * En + j];
    et[i] = v;
    xh[(size_t)b * KG + j] = (bf16)v;
}

// ---------------------------------------------------------------------------
// Attention: scores -> softmax -> beta -> context z. One block per batch elem.
// ---------------------------------------------------------------------------
__global__ __launch_bounds__(256) void attention_kernel(
    const float* __restrict__ vatt,   // [B,N,A]
    const float* __restrict__ q,      // [B,A]
    const float* __restrict__ vw, const float* __restrict__ vb,
    const float* __restrict__ h,      // [B,D] fp32
    const float* __restrict__ betaW, const float* __restrict__ betab,
    const float* __restrict__ fmaps,  // [B,N,ENC] fp32
    float* __restrict__ alphas_out,   // [B,N]
    float* __restrict__ ctx_out,      // [B,ENC]
    bf16* __restrict__ xh,            // z -> xh[:, 512:2560]
    bf16* __restrict__ hz)            // z -> hz[:, 512:2560]
{
    __shared__ float sc[Nn];
    __shared__ float red[256];
    __shared__ float sh[2];           // [0]=1/sum, [1]=beta
    const int b = blockIdx.x, t = threadIdx.x;

    if (t < Nn) {
        const float* va = vatt + ((size_t)b * Nn + t) * An;
        const float* qb = q + (size_t)b * An;
        float s = vb[0];
        for (int a = 0; a < An; ++a) s += vw[a] * tanhf(va[a] + qb[a]);
        sc[t] = s;
    }
    __syncthreads();
    if (t == 0) {
        float mx = sc[0];
        for (int n = 1; n < Nn; ++n) mx = fmaxf(mx, sc[n]);
        float sum = 0.f;
        for (int n = 0; n < Nn; ++n) { float e = __expf(sc[n] - mx); sc[n] = e; sum += e; }
        sh[0] = 1.0f / sum;
    }
    __syncthreads();
    if (t < Nn) {
        float a = sc[t] * sh[0];
        sc[t] = a;
        alphas_out[(size_t)b * Nn + t] = a;
    }
    // beta = sigmoid(h . betaW + betab)
    float p = 0.f;
    for (int d2 = t; d2 < Dn; d2 += 256) p += h[(size_t)b * Dn + d2] * betaW[d2];
    red[t] = p;
    __syncthreads();
    for (int off = 128; off; off >>= 1) {
        if (t < off) red[t] += red[t + off];
        __syncthreads();
    }
    if (t == 0) sh[1] = sigm(red[0] + betab[0]);
    __syncthreads();

    const float beta = sh[1];
    for (int f = t; f < ENCn; f += 256) {
        const float* fb = fmaps + (size_t)b * Nn * ENCn + f;
        float acc = 0.f;
        for (int n = 0; n < Nn; ++n) acc += sc[n] * fb[(size_t)n * ENCn];
        float z = beta * acc;
        ctx_out[(size_t)b * ENCn + f] = z;
        bf16 zb = (bf16)z;
        xh[(size_t)b * KG + En + f] = zb;
        hz[(size_t)b * KO + Dn + f] = zb;
    }
}

// ---------------------------------------------------------------------------
// LSTM elementwise (PyTorch gate order i,f,g,o)
// ---------------------------------------------------------------------------
__global__ void lstm_kernel(const float* __restrict__ gates,  // [B, 4D]
                            float* __restrict__ c, float* __restrict__ h,
                            bf16* __restrict__ xh, bf16* __restrict__ hz) {
    int i = blockIdx.x * blockDim.x + threadIdx.x;   // Bn*Dn
    if (i >= Bn * Dn) return;
    int b = i >> 9, d = i & 511;
    const float* g = gates + (size_t)b * (4 * Dn);
    float ig = sigm(g[d]);
    float fg = sigm(g[Dn + d]);
    float gg = tanhf(g[2 * Dn + d]);
    float og = sigm(g[3 * Dn + d]);
    float cn = fg * c[i] + ig * gg;
    float hn = og * tanhf(cn);
    c[i] = cn;
    h[i] = hn;
    bf16 hb = (bf16)hn;
    xh[(size_t)b * KG + 2560 + d] = hb;   // next step's gates GEMM + q GEMM
    hz[(size_t)b * KO + d]        = hb;   // this step's out-head GEMM
}

// ---------------------------------------------------------------------------
// Host side
// ---------------------------------------------------------------------------
static inline int cdiv(long long a, long long b) { return (int)((a + b - 1) / b); }

template <typename AT, int NT>
static void launch_gemm(hipStream_t st, const AT* A, int lda, const bf16* Bw, int ldb,
                        const float* bias, const float* add, int ldadd,
                        float* C, int ldc, bf16* Cb, int ldcb, int M, int N, int K) {
    int waves  = (M / 16) * (N / (16 * NT));
    int blocks = cdiv(waves, 4);
    wmma_gemm<AT, NT><<<blocks, 128, 0, st>>>(A, lda, Bw, ldb, bias, add, ldadd,
                                              C, ldc, Cb, ldcb, M, N, K);
}

extern "C" void kernel_launch(void* const* d_in, const int* in_sizes, int n_in,
                              void* d_out, int out_size, void* d_ws, size_t ws_size,
                              hipStream_t stream) {
    (void)in_sizes; (void)n_in; (void)out_size; (void)ws_size;

    const float* fmaps  = (const float*)d_in[0];
    const float* fmean  = (const float*)d_in[1];
    const int*   cap    = (const int*)  d_in[2];
    const float* emb    = (const float*)d_in[3];
    const float* ihW    = (const float*)d_in[4];
    const float* ihb    = (const float*)d_in[5];
    const float* icW    = (const float*)d_in[6];
    const float* icb    = (const float*)d_in[7];
    const float* Wih    = (const float*)d_in[8];
    const float* Whh    = (const float*)d_in[9];
    const float* bih    = (const float*)d_in[10];
    const float* bhh    = (const float*)d_in[11];
    const float* W1     = (const float*)d_in[12];
    const float* b1     = (const float*)d_in[13];
    const float* W2     = (const float*)d_in[14];
    const float* b2     = (const float*)d_in[15];
    const float* vw     = (const float*)d_in[16];
    const float* vb     = (const float*)d_in[17];
    const float* betaW  = (const float*)d_in[18];
    const float* betab  = (const float*)d_in[19];
    const float* hidW   = (const float*)d_in[20];
    const float* hidb   = (const float*)d_in[21];
    const float* ctxW   = (const float*)d_in[22];
    const float* ctxb   = (const float*)d_in[23];
    const float* outW   = (const float*)d_in[24];
    const float* outb   = (const float*)d_in[25];

    // output layout: preds [19,64,32000] | ctxs [19,64,2048] | alphas [19,64,196]
    float* out_preds = (float*)d_out;
    float* out_ctx   = out_preds + (size_t)Sn * Bn * Vn;
    float* out_alph  = out_ctx   + (size_t)Sn * Bn * ENCn;

    // ---- workspace bump allocator (~83 MB total) ----
    char*  ws  = (char*)d_ws;
    size_t off = 0;
    auto alloc = [&](size_t bytes) -> void* {
        off = (off + 255) & ~(size_t)255;
        void* p = ws + off;
        off += bytes;
        return p;
    };
    bf16*  outWb = (bf16*) alloc((size_t)Vn * En * 2);       // 32.8 MB (fits L2 across 19 reuses)
    bf16*  Wcat  = (bf16*) alloc((size_t)(4 * Dn) * KG * 2); // 12.6 MB
    bf16*  Wocat = (bf16*) alloc((size_t)En * KO * 2);
    bf16*  W1b   = (bf16*) alloc((size_t)An * ENCn * 2);
    bf16*  W2b   = (bf16*) alloc((size_t)An * Dn * 2);
    bf16*  ihWb  = (bf16*) alloc((size_t)Dn * ENCn * 2);
    bf16*  icWb  = (bf16*) alloc((size_t)Dn * ENCn * 2);
    float* vatt  = (float*)alloc((size_t)Bn * Nn * An * 4);  // 25.7 MB
    float* qf    = (float*)alloc((size_t)Bn * An * 4);
    float* gatesf= (float*)alloc((size_t)Bn * 4 * Dn * 4);
    float* hbuf  = (float*)alloc((size_t)Bn * Dn * 4);
    float* cbuf  = (float*)alloc((size_t)Bn * Dn * 4);
    float* etf   = (float*)alloc((size_t)Bn * En * 4);
    bf16*  xh    = (bf16*) alloc((size_t)Bn * KG * 2);       // [e | z | h]
    bf16*  hz    = (bf16*) alloc((size_t)Bn * KO * 2);       // [h | z]
    bf16*  outbf = (bf16*) alloc((size_t)Bn * En * 2);
    float* bg    = (float*)alloc((size_t)4 * Dn * 4);
    float* bo    = (float*)alloc((size_t)En * 4);

    // ---- one-time setup ----
    cvt_f32_bf16<<<cdiv((size_t)Vn * En, 256), 256, 0, stream>>>(outW, outWb, Vn * En);
    cvt_f32_bf16<<<cdiv((size_t)An * ENCn, 256), 256, 0, stream>>>(W1, W1b, An * ENCn);
    cvt_f32_bf16<<<cdiv((size_t)An * Dn, 256), 256, 0, stream>>>(W2, W2b, An * Dn);
    cvt_f32_bf16<<<cdiv((size_t)Dn * ENCn, 256), 256, 0, stream>>>(ihW, ihWb, Dn * ENCn);
    cvt_f32_bf16<<<cdiv((size_t)Dn * ENCn, 256), 256, 0, stream>>>(icW, icWb, Dn * ENCn);
    concat_gates_w<<<cdiv((size_t)4 * Dn * KG, 256), 256, 0, stream>>>(Wih, Whh, Wcat);
    concat_out_w<<<cdiv((size_t)En * KO, 256), 256, 0, stream>>>(hidW, ctxW, Wocat);
    add_bias<<<cdiv(4 * Dn, 256), 256, 0, stream>>>(bih, bhh, bg, 4 * Dn);
    add_bias<<<cdiv(En, 256), 256, 0, stream>>>(hidb, ctxb, bo, En);

    // values_att = feature_maps @ W1^T + b1   (fp32 A converted on load; big -> NT=4)
    launch_gemm<float, 4>(stream, fmaps, ENCn, W1b, ENCn, b1, nullptr, 0,
                          vatt, An, nullptr, 0, Bn * Nn, An, ENCn);
    // h0/c0 = feature_mean @ init_{h,c}_W^T + b ; h0 also mirrored bf16 into xh[:,2560:]
    launch_gemm<float, 1>(stream, fmean, ENCn, ihWb, ENCn, ihb, nullptr, 0,
                          hbuf, Dn, xh + 2560, KG, Bn, Dn, ENCn);
    launch_gemm<float, 1>(stream, fmean, ENCn, icWb, ENCn, icb, nullptr, 0,
                          cbuf, Dn, nullptr, 0, Bn, Dn, ENCn);

    // ---- 19 sequential decode steps ----
    for (int s = 0; s < Sn; ++s) {
        gather_emb<<<cdiv(Bn * En, 256), 256, 0, stream>>>(cap, emb, s, etf, xh);

        // q = h @ W2^T + b2   (h read as bf16 slice of xh; small -> NT=1 for wave count)
        launch_gemm<bf16, 1>(stream, xh + 2560, KG, W2b, Dn, b2, nullptr, 0,
                             qf, An, nullptr, 0, Bn, An, Dn);

        attention_kernel<<<Bn, 256, 0, stream>>>(
            vatt, qf, vw, vb, hbuf, betaW, betab, fmaps,
            out_alph + (size_t)s * Bn * Nn, out_ctx + (size_t)s * Bn * ENCn, xh, hz);

        // gates = [e|z|h] @ [W_ih|W_hh]^T + (b_ih+b_hh)   (NT=2 -> 256 waves)
        launch_gemm<bf16, 2>(stream, xh, KG, Wcat, KG, bg, nullptr, 0,
                             gatesf, 4 * Dn, nullptr, 0, Bn, 4 * Dn, KG);

        lstm_kernel<<<cdiv(Bn * Dn, 256), 256, 0, stream>>>(gatesf, cbuf, hbuf, xh, hz);

        // out = e_t + [h|z] @ [hid_W|ctx_W]^T + (hid_b+ctx_b)  -> bf16 only
        launch_gemm<bf16, 1>(stream, hz, KO, Wocat, KO, bo, etf, En,
                             nullptr, 0, outbf, En, Bn, En, KO);

        // preds = out @ out_W^T + out_b  -> straight into d_out  (NT=4 -> 2000 waves)
        launch_gemm<bf16, 4>(stream, outbf, En, outWb, En, outb, nullptr, 0,
                             out_preds + (size_t)s * Bn * Vn, Vn, nullptr, 0, Bn, Vn, En);
    }
}
